// DiffiTAttention_51075751084659
// MI455X (gfx1250) — matrix operand order repaired and checked by reading, not compiled, not run on hardware
//
#include <hip/hip_runtime.h>
#include <hip/hip_bf16.h>

// ---------------------------------------------------------------------------
// DiffiT attention block for gfx1250 (MI455X): bf16 WMMA GEMMs + flash attn.
// Weights pre-converted to bf16 AND pre-transposed to N-major so both LDS
// tiles fill with contiguous async b128 copies and both fragment types read
// two contiguous 16B runs per lane (ds_load_b128).
// ---------------------------------------------------------------------------

typedef __attribute__((ext_vector_type(16))) __bf16 v16bf;
typedef __attribute__((ext_vector_type(4)))  __bf16 v4bf;
typedef __attribute__((ext_vector_type(8)))  float  v8f;
typedef __attribute__((ext_vector_type(4)))  int    v4i;

#define DEV __device__ __forceinline__

// ---- async global->LDS (CDNA5 GLOBAL_LOAD_ASYNC_TO_LDS_B128, ASYNCcnt) ----
#if defined(__HIP_DEVICE_COMPILE__) && \
    __has_builtin(__builtin_amdgcn_global_load_async_to_lds_b128)
#define HAVE_ASYNC_LDS 1
DEV void async_copy_b128(__bf16* lds_dst, const __bf16* g_src) {
  __builtin_amdgcn_global_load_async_to_lds_b128(
      (v4i __attribute__((address_space(1)))*)g_src,
      (v4i __attribute__((address_space(3)))*)lds_dst, 0, 0);
}
DEV void wait_async0() {
#if __has_builtin(__builtin_amdgcn_s_wait_asynccnt)
  __builtin_amdgcn_s_wait_asynccnt(0);
#else
  asm volatile("s_wait_asynccnt 0" ::: "memory");
#endif
}
#else
#define HAVE_ASYNC_LDS 0
#endif

DEV v8f wmma_bf16(v16bf a, v16bf b, v8f c) {
  // D = A(16x32 bf16) * B(32x16 bf16) + C(16x16 f32)
  return __builtin_amdgcn_wmma_f32_16x16x32_bf16(
      /*neg_a=*/false, a, /*neg_b=*/false, b,
      /*c_mod=*/(short)0, c, /*reuse_a=*/false, /*reuse_b=*/false);
}

// A fragment, row-major bf16 source: logical element (m, k) at p[m*ld + k].
// ISA 7.12.2 16-bit A layout: lanes 0-15 M=lane; VGPR r holds K pairs
// {2r+8*half} (r<4) / {2r+8+8*half} (r>=4)  -> two contiguous 16B runs.
DEV v16bf frag_a_bf(const __bf16* p, int ld, int lane) {
  int half = (lane >> 4) & 1, m = lane & 15;
  v16bf a;
#pragma unroll
  for (int r = 0; r < 8; ++r) {
    int k = (r < 4) ? (2 * r + 8 * half) : (2 * r + 8 + 8 * half);
    a[2 * r]     = p[m * ld + k];
    a[2 * r + 1] = p[m * ld + k + 1];
  }
  return a;
}

// B fragment from an N-major source: logical element (kk, n) at p[n*ld + kk].
// 16-bit B layout matching A: VGPR r holds K pairs {2r+8*half} (r<4) /
// {2r+8+8*half} (r>=4); N = lane%16. Per lane: two contiguous 16B runs.
DEV v16bf frag_b_nmajor(const __bf16* p, int ld, int lane) {
  int half = (lane >> 4) & 1, n = lane & 15;
  v16bf b;
#pragma unroll
  for (int r = 0; r < 8; ++r) {
    int kk = (r < 4) ? (2 * r + 8 * half) : (2 * r + 8 + 8 * half);
    b[2 * r]     = p[n * ld + kk];
    b[2 * r + 1] = p[n * ld + kk + 1];
  }
  return b;
}

// ---------------------------------------------------------------------------
// Kernel 0a: f32 -> bf16 bulk conversion (one float4 / thread)
// ---------------------------------------------------------------------------
__global__ void cvt_bf16_kernel(const float* __restrict__ src,
                                __bf16* __restrict__ dst) {
  size_t i = (size_t)blockIdx.x * blockDim.x + threadIdx.x;
  float4 v = ((const float4*)src)[i];
  v4bf o = {(__bf16)v.x, (__bf16)v.y, (__bf16)v.z, (__bf16)v.w};
  ((v4bf*)dst)[i] = o;
}

// ---------------------------------------------------------------------------
// Kernel 0b: f32 [K][N] -> bf16 [N][K] transpose-convert (32x32 LDS tiles)
// ---------------------------------------------------------------------------
__global__ __launch_bounds__(256) void cvt_transpose_bf16_kernel(
    const float* __restrict__ src, __bf16* __restrict__ dst, int K, int N) {
  __shared__ float tile[32][33];
  int k0 = blockIdx.y * 32, n0 = blockIdx.x * 32;
  int tx = threadIdx.x & 31, ty = threadIdx.x >> 5;  // ty = 0..7
#pragma unroll
  for (int i = 0; i < 32; i += 8)
    tile[ty + i][tx] = src[(size_t)(k0 + ty + i) * N + n0 + tx];
  __syncthreads();
#pragma unroll
  for (int i = 0; i < 32; i += 8)
    dst[(size_t)(n0 + ty + i) * K + k0 + tx] = (__bf16)tile[tx][ty + i];
}

// ---------------------------------------------------------------------------
// Kernel 1: tvec[b][j] = temb[b] . Wt[:,j] + bt[j] + bqkv[j]   (tiny GEMM)
// ---------------------------------------------------------------------------
__global__ void temb_proj_kernel(const float* __restrict__ temb,
                                 const float* __restrict__ Wt,
                                 const float* __restrict__ bt,
                                 const float* __restrict__ bqkv,
                                 float* __restrict__ tvec) {
  int j = blockIdx.x * blockDim.x + threadIdx.x;  // 0..3071
  int b = blockIdx.y;                             // 0..7
  float acc = 0.f;
  const float* tb = temb + b * 1024;
  for (int t = 0; t < 1024; ++t) acc += tb[t] * Wt[(size_t)t * 3072 + j];
  tvec[b * 3072 + j] = acc + bt[j] + bqkv[j];
}

// ---------------------------------------------------------------------------
// Kernel 2/5: D[M,N] = A[M,K] @ Bt[N,K]^T (+bias[N]) (+addvec[row/div][N])
// bf16 A row-major, Bt N-major. 128x128x32 tiles, 256 threads = 8 waves,
// each wave a 32x64 patch (8 wmma / K-step). Both LDS tiles are [128][32]
// bf16 (8KB each) filled with identical contiguous async b128 copies.
// ---------------------------------------------------------------------------
__global__ __launch_bounds__(256) void gemm_wmma_kernel(
    const __bf16* __restrict__ A, const __bf16* __restrict__ Bt,
    const float* __restrict__ bias, const float* __restrict__ addvec,
    int addvec_rowdiv, int addvec_ld, float* __restrict__ D,
    int M, int N, int K) {
  __shared__ __bf16 As[128 * 32];
  __shared__ __bf16 BsT[128 * 32];

  const int tid = threadIdx.x, lane = tid & 31, wave = tid >> 5;
  const int bm = blockIdx.y * 128, bn = blockIdx.x * 128;
  const int wm = (wave & 3) * 32;   // 4 waves along M
  const int wn = (wave >> 2) * 64;  // 2 waves along N

  v8f acc[2][4];
#pragma unroll
  for (int mi = 0; mi < 2; ++mi)
#pragma unroll
    for (int ni = 0; ni < 4; ++ni) { v8f z = {}; acc[mi][ni] = z; }

  for (int kt = 0; kt < K; kt += 32) {
    // tile fill: 512 chunks of 8 bf16 (16B) per tile, 2 chunks/thread
#pragma unroll
    for (int i = 0; i < 2; ++i) {
      int idx = tid + i * 256;                  // 0..511
      int row = idx >> 2, k8 = (idx & 3) * 8;
      const __bf16* ga = &A[(size_t)(bm + row) * K + kt + k8];
      const __bf16* gb = &Bt[(size_t)(bn + row) * K + kt + k8];
#if HAVE_ASYNC_LDS
      async_copy_b128(&As[idx * 8], ga);
      async_copy_b128(&BsT[idx * 8], gb);
#else
      *(uint4*)&As[idx * 8] = *(const uint4*)ga;
      *(uint4*)&BsT[idx * 8] = *(const uint4*)gb;
#endif
    }
    if (kt + 32 < K) {  // global_prefetch_b8 of next tiles
      __builtin_prefetch(&A[(size_t)(bm + (tid >> 1)) * K + kt + 32], 0, 1);
      __builtin_prefetch(&Bt[(size_t)(bn + (tid >> 1)) * K + kt + 32], 0, 1);
    }
#if HAVE_ASYNC_LDS
    wait_async0();
#endif
    __syncthreads();

    v16bf af0 = frag_a_bf(&As[(wm + 0) * 32], 32, lane);
    v16bf af1 = frag_a_bf(&As[(wm + 16) * 32], 32, lane);
#pragma unroll
    for (int ni = 0; ni < 4; ++ni) {
      v16bf bfrag = frag_b_nmajor(&BsT[(wn + ni * 16) * 32], 32, lane);
      acc[0][ni] = wmma_bf16(af0, bfrag, acc[0][ni]);
      acc[1][ni] = wmma_bf16(af1, bfrag, acc[1][ni]);
    }
    __syncthreads();
  }

  const int half = lane >> 4, n = lane & 15;
#pragma unroll
  for (int mi = 0; mi < 2; ++mi)
#pragma unroll
    for (int ni = 0; ni < 4; ++ni)
#pragma unroll
      for (int r = 0; r < 8; ++r) {
        int row = bm + wm + mi * 16 + half * 8 + r;
        int col = bn + wn + ni * 16 + n;
        float v = acc[mi][ni][r];
        if (bias) v += bias[col];
        if (addvec)
          v += addvec[(size_t)(row / addvec_rowdiv) * addvec_ld + col];
        D[(size_t)row * N + col] = v;
      }
}

// ---------------------------------------------------------------------------
// Kernel 3: fused per-head RMSNorm + axial RoPE; emit bf16 q/k [B,H,N,64]
// and v transposed [B,H,64,N]. One wave per (b,h,n) row; lane l holds
// d=l and d=l+32; rotate_half partner within each 32-half is shfl_xor(.,16).
// ---------------------------------------------------------------------------
__global__ __launch_bounds__(256) void qk_prep_kernel(
    const float* __restrict__ qkv, const float* __restrict__ cos_y,
    const float* __restrict__ sin_y, const float* __restrict__ cos_x,
    const float* __restrict__ sin_x, const float* __restrict__ qn_w,
    const float* __restrict__ kn_w, __bf16* __restrict__ qb,
    __bf16* __restrict__ kb, __bf16* __restrict__ vt) {
  const int Nn = 1024, Hh = 16, HD = 64, C3 = 3072;
  int wid = blockIdx.x * (blockDim.x >> 5) + (threadIdx.x >> 5);
  int lane = threadIdx.x & 31;
  int b = wid >> 14;  // /(16*1024)
  int h = (wid >> 10) & 15;
  int n = wid & 1023;

  size_t base = (size_t)(b * Nn + n) * C3 + h * HD;
  float q0 = qkv[base + lane],        q1 = qkv[base + lane + 32];
  float k0 = qkv[base + 1024 + lane], k1 = qkv[base + 1024 + lane + 32];
  float v0 = qkv[base + 2048 + lane], v1 = qkv[base + 2048 + lane + 32];

  float sq = q0 * q0 + q1 * q1, sk = k0 * k0 + k1 * k1;
#pragma unroll
  for (int off = 16; off; off >>= 1) {
    sq += __shfl_xor(sq, off, 32);
    sk += __shfl_xor(sk, off, 32);
  }
  float rq = rsqrtf(sq * (1.f / 64.f) + 1e-6f);
  float rk = rsqrtf(sk * (1.f / 64.f) + 1e-6f);
  q0 = q0 * rq * qn_w[lane];  q1 = q1 * rq * qn_w[lane + 32];
  k0 = k0 * rk * kn_w[lane];  k1 = k1 * rk * kn_w[lane + 32];

  float sgn = (lane < 16) ? -1.f : 1.f;
  float rq0 = sgn * __shfl_xor(q0, 16, 32);
  float rq1 = sgn * __shfl_xor(q1, 16, 32);
  float rk0 = sgn * __shfl_xor(k0, 16, 32);
  float rk1 = sgn * __shfl_xor(k1, 16, 32);

  int ci = n * 32 + lane;  // [1,1,N,32]
  float qy = q0 * cos_y[ci] + rq0 * sin_y[ci];
  float qx = q1 * cos_x[ci] + rq1 * sin_x[ci];
  float ky = k0 * cos_y[ci] + rk0 * sin_y[ci];
  float kx = k1 * cos_x[ci] + rk1 * sin_x[ci];

  size_t ob = (((size_t)b * Hh + h) * Nn + n) * HD;
  qb[ob + lane] = (__bf16)qy;  qb[ob + lane + 32] = (__bf16)qx;
  kb[ob + lane] = (__bf16)ky;  kb[ob + lane + 32] = (__bf16)kx;
  // v transposed: vt[b,h,d,n]
  size_t vtb = (((size_t)b * Hh + h) * HD) * Nn + n;
  vt[vtb + (size_t)lane * Nn]        = (__bf16)v0;
  vt[vtb + (size_t)(lane + 32) * Nn] = (__bf16)v1;
}

// ---------------------------------------------------------------------------
// Kernel 4: flash-style attention. Block = 4 waves; wave owns 16 q rows.
// Per 64-key block: 8 WMMA (QK^T), online softmax via 16-lane-group shfl_xor
// reductions, P staged bf16 through per-wave LDS (C->A relayout), 8 WMMA (PV
// against transposed V). All B fragments are N-major => two contiguous 16B runs/lane.
// Output bf16 [B,N,H*hd].
// ---------------------------------------------------------------------------
__global__ __launch_bounds__(128) void attention_wmma_kernel(
    const __bf16* __restrict__ qb, const __bf16* __restrict__ kbm,
    const __bf16* __restrict__ vt, __bf16* __restrict__ out) {
  const int Nn = 1024, HD = 64, Hh = 16;
  const int bh = blockIdx.y;  // 0..127 (b*16+h)
  const int qt = blockIdx.x;  // 0..15
  const int wave = threadIdx.x >> 5, lane = threadIdx.x & 31;
  const int half = lane >> 4, nl = lane & 15;

  const __bf16* qp = qb  + (size_t)bh * Nn * HD;
  const __bf16* kp = kbm + (size_t)bh * Nn * HD;
  const __bf16* vp = vt  + (size_t)bh * HD * Nn;  // [d][n]
  const int qrow0 = qt * 64 + wave * 16;

  v16bf aq0 = frag_a_bf(qp + (size_t)qrow0 * HD, HD, lane);       // K=0..31
  v16bf aq1 = frag_a_bf(qp + (size_t)qrow0 * HD + 32, HD, lane);  // K=32..63

  v8f o[4];
#pragma unroll
  for (int dn = 0; dn < 4; ++dn) { v8f z = {}; o[dn] = z; }
  float mrow[8], lrow[8];
#pragma unroll
  for (int r = 0; r < 8; ++r) { mrow[r] = -3.0e38f; lrow[r] = 0.f; }

  __shared__ __bf16 pbuf[4][16 * 64];
  __bf16* pw = &pbuf[wave][0];

  for (int kb0 = 0; kb0 < Nn; kb0 += 64) {
    // ---- scores: 4 sub-tiles of 16 keys -------------------------------
    v8f s[4];
#pragma unroll
    for (int j = 0; j < 4; ++j) {
      // k^T fragment: element (kk, n) at kp[(key n)*64 + kk]  (N-major)
      v16bf bk0 = frag_b_nmajor(kp + (size_t)(kb0 + j * 16) * HD, HD, lane);
      v16bf bk1 =
          frag_b_nmajor(kp + (size_t)(kb0 + j * 16) * HD + 32, HD, lane);
      v8f t = {};
      t = wmma_bf16(aq0, bk0, t);
      t = wmma_bf16(aq1, bk1, t);
#pragma unroll
      for (int r = 0; r < 8; ++r) t[r] *= 0.125f;  // 1/sqrt(64)
      s[j] = t;
    }
    // ---- online softmax stats (rows live in 16-lane groups) -----------
    float mnew[8], scl[8];
#pragma unroll
    for (int r = 0; r < 8; ++r) {
      float v = fmaxf(fmaxf(s[0][r], s[1][r]), fmaxf(s[2][r], s[3][r]));
#pragma unroll
      for (int off = 8; off; off >>= 1) v = fmaxf(v, __shfl_xor(v, off, 32));
      mnew[r] = fmaxf(mrow[r], v);
      scl[r] = __expf(mrow[r] - mnew[r]);
      mrow[r] = mnew[r];
    }
    float rs[8];
#pragma unroll
    for (int r = 0; r < 8; ++r) rs[r] = 0.f;
#pragma unroll
    for (int j = 0; j < 4; ++j)
#pragma unroll
      for (int r = 0; r < 8; ++r) {
        float p = __expf(s[j][r] - mnew[r]);
        rs[r] += p;
        pw[(half * 8 + r) * 64 + j * 16 + nl] = (__bf16)p;
      }
#pragma unroll
    for (int r = 0; r < 8; ++r) {
      float t = rs[r];
#pragma unroll
      for (int off = 8; off; off >>= 1) t += __shfl_xor(t, off, 32);
      lrow[r] = lrow[r] * scl[r] + t;
#pragma unroll
      for (int dn = 0; dn < 4; ++dn) o[dn][r] *= scl[r];
    }
    // ---- P @ V (A from per-wave LDS; B from transposed V, N-major) ----
    v16bf ap0 = frag_a_bf(pw, 64, lane);
    v16bf ap1 = frag_a_bf(pw + 32, 64, lane);
#pragma unroll
    for (int dn = 0; dn < 4; ++dn) {
      // v fragment: element (kk, n=d) at vp[(dn*16+n)*Nn + kb0 + kk]
      v16bf bv0 =
          frag_b_nmajor(vp + (size_t)(dn * 16) * Nn + kb0, Nn, lane);
      v16bf bv1 =
          frag_b_nmajor(vp + (size_t)(dn * 16) * Nn + kb0 + 32, Nn, lane);
      o[dn] = wmma_bf16(ap0, bv0, o[dn]);
      o[dn] = wmma_bf16(ap1, bv1, o[dn]);
    }
  }

  // ---- epilogue: out[b, n, h*64 + d] = o / l (bf16) -------------------
  const int b = bh >> 4, h = bh & 15;
#pragma unroll
  for (int dn = 0; dn < 4; ++dn)
#pragma unroll
    for (int r = 0; r < 8; ++r) {
      int row = qrow0 + half * 8 + r;
      float val = o[dn][r] / lrow[r];
      out[(((size_t)b * Nn + row) * Hh + h) * HD + dn * 16 + nl] =
          (__bf16)val;
    }
}

// ---------------------------------------------------------------------------
// Host driver
// ---------------------------------------------------------------------------
extern "C" void kernel_launch(void* const* d_in, const int* in_sizes, int n_in,
                              void* d_out, int out_size, void* d_ws,
                              size_t ws_size, hipStream_t stream) {
  const float* x     = (const float*)d_in[0];   // [8,1024,1024]
  const float* temb  = (const float*)d_in[1];   // [8,1024]
  const float* cos_y = (const float*)d_in[2];   // [1,1,1024,32]
  const float* sin_y = (const float*)d_in[3];
  const float* cos_x = (const float*)d_in[4];
  const float* sin_x = (const float*)d_in[5];
  const float* Wqkv  = (const float*)d_in[6];   // [1024,3072]
  const float* bqkv  = (const float*)d_in[7];   // [3072]
  const float* Wt    = (const float*)d_in[8];   // [1024,3072]
  const float* bt    = (const float*)d_in[9];   // [3072]
  const float* Wp    = (const float*)d_in[10];  // [1024,1024]
  const float* bp    = (const float*)d_in[11];  // [1024]
  const float* qn_w  = (const float*)d_in[12];  // [64]
  const float* kn_w  = (const float*)d_in[13];  // [64]
  float* outp = (float*)d_out;                  // [8,1024,1024]

  const int B = 8, N = 1024, C = 1024, H = 16, HD = 64;
  const int M = B * N;   // 8192
  const int C3 = 3 * C;  // 3072

  // workspace carve-up
  char* ws = (char*)d_ws;
  float*  tvec    = (float*)ws;  ws += (size_t)B * C3 * 4;            // 96 KB
  float*  qkv     = (float*)ws;  ws += (size_t)M * C3 * 4;            // 96 MB
  __bf16* xb      = (__bf16*)ws; ws += (size_t)M * C * 2;             // 16 MB
  __bf16* wqkv_t  = (__bf16*)ws; ws += (size_t)C * C3 * 2;            //  6 MB
  __bf16* wp_t    = (__bf16*)ws; ws += (size_t)C * C * 2;             //  2 MB
  __bf16* qb      = (__bf16*)ws; ws += (size_t)B * H * N * HD * 2;    // 16 MB
  __bf16* kb      = (__bf16*)ws; ws += (size_t)B * H * N * HD * 2;    // 16 MB
  __bf16* vtb     = (__bf16*)ws; ws += (size_t)B * H * HD * N * 2;    // 16 MB
  __bf16* attnb   = (__bf16*)ws; ws += (size_t)M * C * 2;             // 16 MB

  // 0) x -> bf16; weights -> bf16 transposed ([N][K])
  cvt_bf16_kernel<<<(M * C) / (256 * 4), 256, 0, stream>>>(x, xb);
  cvt_transpose_bf16_kernel<<<dim3(C3 / 32, C / 32), 256, 0, stream>>>(
      Wqkv, wqkv_t, C, C3);
  cvt_transpose_bf16_kernel<<<dim3(C / 32, C / 32), 256, 0, stream>>>(
      Wp, wp_t, C, C);

  // 1) time-conditioning vector: tvec = temb @ Wt + bt + bqkv
  temb_proj_kernel<<<dim3(C3 / 256, B), 256, 0, stream>>>(temb, Wt, bt, bqkv,
                                                          tvec);

  // 2) qkv = x @ Wqkv + tvec[b]  (bqkv folded into tvec)
  gemm_wmma_kernel<<<dim3(C3 / 128, M / 128), 256, 0, stream>>>(
      xb, wqkv_t, nullptr, tvec, /*rowdiv=*/N, /*ld=*/C3, qkv, M, C3, C);

  // 3) RMSNorm + RoPE -> bf16 q/k [B,H,N,64], v^T [B,H,64,N]
  qk_prep_kernel<<<dim3((B * H * N) / 8), 256, 0, stream>>>(
      qkv, cos_y, sin_y, cos_x, sin_x, qn_w, kn_w, qb, kb, vtb);

  // 4) flash attention -> attnb [B,N,C] bf16
  attention_wmma_kernel<<<dim3(N / 64, B * H), 128, 0, stream>>>(qb, kb, vtb,
                                                                 attnb);

  // 5) out = attn @ Wp + bp
  gemm_wmma_kernel<<<dim3(C / 128, M / 128), 256, 0, stream>>>(
      attnb, wp_t, bp, nullptr, /*rowdiv=*/1, /*ld=*/0, outp, M, C, C);
}
